// CostVolume_3169685865265
// MI455X (gfx1250) — compile-verified, matrix operand order checked by reading, not run on hardware
//
#include <hip/hip_runtime.h>
#include <hip/hip_bf16.h>

typedef _Float16 v16h __attribute__((ext_vector_type(16)));
typedef float    v8f  __attribute__((ext_vector_type(8)));
typedef unsigned int u32x4 __attribute__((ext_vector_type(4)));
typedef int          i32x8 __attribute__((ext_vector_type(8)));
typedef int          i32x4 __attribute__((ext_vector_type(4)));

#define B_  2
#define S_  4096
#define N_  4096
#define KQ  32
#define KS  16

// ---------------------------------------------------------------------------
// Fold BN into scale/bias, convert weights to f16 with zero-padded cin.
// ---------------------------------------------------------------------------
__global__ void pack_layer(const float* __restrict__ W,
                           const float* __restrict__ beta,
                           const float* __restrict__ gamma,
                           const float* __restrict__ mean,
                           const float* __restrict__ var,
                           int cout, int cin, int cinpad,
                           _Float16* __restrict__ Wh,
                           float* __restrict__ scale,
                           float* __restrict__ bias) {
  int i = blockIdx.x * blockDim.x + threadIdx.x;
  int total = cout * cinpad;
  if (i < total) {
    int co = i / cinpad, ci = i % cinpad;
    Wh[i] = (ci < cin) ? (_Float16)W[co * cin + ci] : (_Float16)0.0f;
  }
  if (i < cout) {
    float inv = rsqrtf(var[i] + 1e-5f);
    float g = gamma[i] * inv;
    scale[i] = g;
    bias[i] = beta[i] - mean[i] * g;
  }
}

// ---------------------------------------------------------------------------
// TDM: DMA one [3 rows x 256 cols] fp32 tile of a [3][Nd] tensor into LDS.
// D# packing per CDNA5 ISA (2-D tensor, groups 2/3 = 0):
//   group0: count=1 | lds_addr | global_addr[31:0] | global_addr[56:32], type=2
//   group1: data_size=4B; tensor_dim0=Nd; tensor_dim1=3; tile_dim0=256;
//           tile_dim1=3; tensor_dim0_stride=Nd
// Issued by one wave per block; tracked with TENSORcnt.
// 6-arg builtin flavor (clang-23 / therock headers).
// ---------------------------------------------------------------------------
__device__ __forceinline__ void tdm_load_tile(const float* gsrc,
                                              unsigned lds_off, int nd) {
  unsigned long long ga = (unsigned long long)(size_t)gsrc;
  u32x4 g0;
  g0[0] = 1u;                                             // count=1
  g0[1] = lds_off;                                        // lds_addr
  g0[2] = (unsigned)(ga & 0xffffffffull);                 // global_addr lo
  g0[3] = (unsigned)((ga >> 32) & 0x01ffffffull) | (2u << 30);  // hi | type=2
  i32x8 g1;
  g1[0] = (2 << 16);                    // wg_mask=0, data_size=2 (4 bytes)
  g1[1] = (int)((nd & 0xffff) << 16);   // bits 63:48 = tensor_dim0[15:0]
  g1[2] = (int)(((unsigned)nd >> 16) & 0xffffu) | (3 << 16); // dim0 hi | dim1 lo
  g1[3] = (256 << 16);                  // tensor_dim1 hi=0 | tile_dim0=256
  g1[4] = 3;                            // tile_dim1=3, tile_dim2=0
  g1[5] = nd;                           // tensor_dim0_stride[31:0]
  g1[6] = 0;                            // stride hi | dim1_stride lo (unused 2D)
  g1[7] = 0;
  i32x4 z4 = {0, 0, 0, 0};
  i32x8 z8 = {0, 0, 0, 0, 0, 0, 0, 0};
  __builtin_amdgcn_tensor_load_to_lds(g0, g1, z4, z4, z8, 0);
}

// ---------------------------------------------------------------------------
// KNN: one thread per query. DB tiles are DMA'd into LDS by the Tensor Data
// Mover with double buffering: tile t+1 is in flight while tile t feeds the
// top-K insertion loop.  s_wait_tensorcnt 1 == "previous tile has landed".
// ---------------------------------------------------------------------------
template <int K>
__global__ void __launch_bounds__(256) knn_kernel(
    const float* __restrict__ q_xyz,   // [B][3][Sq]
    const float* __restrict__ d_xyz,   // [B][3][Nd]
    int* __restrict__ idx_out,         // [B][Sq][K]
    int Sq, int Nd) {
  __shared__ float tiles[2][3 * 256];  // [buf][row(y)][col(x)]
  __shared__ float sn[256];
  int b = blockIdx.y;
  int s = blockIdx.x * 256 + threadIdx.x;
  int wave = threadIdx.x >> 5;
  const float* qb = q_xyz + (size_t)b * 3 * Sq;
  const float* db = d_xyz + (size_t)b * 3 * Nd;
  float qx = qb[s], qy = qb[Sq + s], qz = qb[2 * Sq + s];
  float qq = qx * qx + qy * qy + qz * qz;

  unsigned lds0 = (unsigned)(size_t)(void*)&tiles[0][0];
  unsigned lds1 = (unsigned)(size_t)(void*)&tiles[1][0];

  float bd[K];
  int   bi[K];
#pragma unroll
  for (int k = 0; k < K; ++k) { bd[k] = 3.4e38f; bi[k] = 0; }

  if (wave == 0) tdm_load_tile(db, lds0, Nd);  // prime tile 0

  int cur = 0;
  for (int t = 0; t < Nd; t += 256, cur ^= 1) {
    __syncthreads();  // everyone done reading the buffer we are about to fill
    if (wave == 0) {
      if (t + 256 < Nd) {
        tdm_load_tile(db + (t + 256), cur ? lds0 : lds1, Nd);
        __builtin_amdgcn_s_wait_tensorcnt(1);  // tile t has landed
      } else {
        __builtin_amdgcn_s_wait_tensorcnt(0);
      }
    }
    __syncthreads();  // tile t visible to all waves

    const float* tx = &tiles[cur][0];
    const float* ty = &tiles[cur][256];
    const float* tz = &tiles[cur][512];
    {
      float x = tx[threadIdx.x], y = ty[threadIdx.x], z = tz[threadIdx.x];
      sn[threadIdx.x] = x * x + y * y + z * z;
    }
    __syncthreads();

    for (int jj = 0; jj < 256; ++jj) {
      float d2 = qq - 2.0f * (qx * tx[jj] + qy * ty[jj] + qz * tz[jj]) + sn[jj];
      if (d2 < bd[K - 1]) {
        int p = K - 1;
        while (p > 0 && bd[p - 1] > d2) {
          bd[p] = bd[p - 1]; bi[p] = bi[p - 1]; --p;
        }
        bd[p] = d2; bi[p] = t + jj;
      }
    }
  }
  int* o = idx_out + ((size_t)b * Sq + s) * K;
#pragma unroll
  for (int k = 0; k < K; ++k) o[k] = bi[k];
}

// ---------------------------------------------------------------------------
// One MLP layer via v_wmma_f32_16x16x32_f16.
//   A = weights [cout][cinpad] f16 (global, tiny, L2-resident)
//   B = activations in LDS, row-major [m][cin] f16
// Fragment packing follows the CDNA5 ISA 16-bit A/B layout:
//   lane l: m/n = l&15, g = l>>4; VGPR j holds k = (j>>2)*16 + g*8 + (j&3)*2 (+1)
// C/D layout: VGPR r -> cout_local = r + 8*g, column = lane&15.
// ---------------------------------------------------------------------------
__device__ __forceinline__ void mlp_layer(
    int wave, int lane,
    const _Float16* __restrict__ Wh,
    const float* __restrict__ scale, const float* __restrict__ bias,
    const _Float16* __restrict__ Xlds, int xstride,
    _Float16* Ylds, int ystride, int ycolofs,
    float* Yf, int fstride,
    int cout, int chunks, int ntiles) {
  const int cinpad = chunks * 32;
  const int ct = wave;
  if (ct * 16 < cout) {
    const int n = lane & 15, g = lane >> 4;
    const _Float16* wrow = Wh + (size_t)(ct * 16 + n) * cinpad;  // A: m = lane&15
    for (int nt = 0; nt < ntiles; ++nt) {
      const _Float16* xrow = Xlds + (size_t)(nt * 16 + n) * xstride;
      v8f acc = {};
      for (int ch = 0; ch < chunks; ++ch) {
        v16h a, bf;
#pragma unroll
        for (int j = 0; j < 8; ++j) {
          int k0 = ch * 32 + (j >> 2) * 16 + g * 8 + (j & 3) * 2;
          a[2 * j]      = wrow[k0];
          a[2 * j + 1]  = wrow[k0 + 1];
          bf[2 * j]     = xrow[k0];
          bf[2 * j + 1] = xrow[k0 + 1];
        }
        acc = __builtin_amdgcn_wmma_f32_16x16x32_f16(
            false, a, false, bf, (short)0, acc, false, false);
      }
#pragma unroll
      for (int r = 0; r < 8; ++r) {
        int co = ct * 16 + r + 8 * g;
        float v = fmaxf(acc[r] * scale[co] + bias[co], 0.0f);  // BN + ReLU
        int m = nt * 16 + n;
        if (Ylds) Ylds[(size_t)m * ystride + ycolofs + co] = (_Float16)v;
        if (Yf)   Yf[(size_t)co * fstride + m] = v;
      }
    }
  }
  __syncthreads();
}

// ---------------------------------------------------------------------------
// Stage 1: one block per (b,s). Gather K=32 neighbors, run
// mlp1 (138->128->64->64), xyz1 (10->64), mlp2 (128->128->64),
// softmax over K, weighted sum -> pi_feat1[b][64][s].
// ---------------------------------------------------------------------------
__global__ void __launch_bounds__(256) stage1_kernel(
    const float* __restrict__ warped_xyz,     // [B][3][S]
    const float* __restrict__ warped_points,  // [B][64][S]
    const float* __restrict__ f2_xyz,         // [B][3][N]
    const float* __restrict__ f2_points,      // [B][64][N]
    const int*   __restrict__ idx_q,          // [B][S][32]
    const _Float16* Wm1a, const float* s1a, const float* b1a,
    const _Float16* Wm1b, const float* s1b, const float* b1b,
    const _Float16* Wm1c, const float* s1c, const float* b1c,
    const _Float16* Wx1,  const float* sx1, const float* bx1,
    const _Float16* Wm2a, const float* s2a, const float* b2a,
    const _Float16* Wm2b, const float* s2b, const float* b2b,
    float* __restrict__ pi_feat1)             // [B][64][S]
{
  __shared__ _Float16 Xin[32 * 160];  // input: 10 xyz-feat + 128 pts-feat, pad->160
  __shared__ _Float16 T1[32 * 128];
  __shared__ _Float16 T2[32 * 128];
  __shared__ float    F1[64 * 32];    // feat1 fp32 for the weighted sum
  __shared__ float    WQ[64 * 32];    // mlp2 output (pre-softmax)
  __shared__ int      nidx[32];
  __shared__ float    nx[32], ny[32], nz[32];

  const int b = blockIdx.y, s = blockIdx.x;
  const int tid = threadIdx.x, lane = tid & 31, wave = tid >> 5;

  if (tid < 32) {
    int j = idx_q[((size_t)b * S_ + s) * 32 + tid];
    nidx[tid] = j;
    nx[tid] = f2_xyz[(size_t)b * 3 * N_ + j];
    ny[tid] = f2_xyz[(size_t)b * 3 * N_ + N_ + j];
    nz[tid] = f2_xyz[(size_t)b * 3 * N_ + 2 * N_ + j];
  }
  __syncthreads();

  const float qx = warped_xyz[(size_t)b * 3 * S_ + s];
  const float qy = warped_xyz[(size_t)b * 3 * S_ + S_ + s];
  const float qz = warped_xyz[(size_t)b * 3 * S_ + 2 * S_ + s];

  for (int e = tid; e < 32 * 160; e += 256) {
    int k = e / 160, c = e % 160;
    float v = 0.0f;
    if (c < 3)        v = (c == 0 ? qx : (c == 1 ? qy : qz));
    else if (c < 6)   v = (c == 3 ? nx[k] : (c == 4 ? ny[k] : nz[k]));
    else if (c < 9)   v = (c == 6 ? nx[k] - qx : (c == 7 ? ny[k] - qy : nz[k] - qz));
    else if (c == 9) {
      float dx = nx[k] - qx, dy = ny[k] - qy, dz = nz[k] - qz;
      v = sqrtf(dx * dx + dy * dy + dz * dz + 1e-20f);
    } else if (c < 74)  v = warped_points[((size_t)b * 64 + (c - 10)) * S_ + s];
    else if (c < 138)   v = f2_points[((size_t)b * 64 + (c - 74)) * N_ + nidx[k]];
    Xin[k * 160 + c] = (_Float16)v;
  }
  __syncthreads();

  // mlp1: 138(pad160) -> 128 -> 64 -> 64  (feat1 into T1 cols 64..127 + F1 f32)
  mlp_layer(wave, lane, Wm1a, s1a, b1a, Xin, 160, T1, 128, 0,  nullptr, 0, 128, 5, 2);
  mlp_layer(wave, lane, Wm1b, s1b, b1b, T1, 128, T2, 128, 0,  nullptr, 0,  64, 4, 2);
  mlp_layer(wave, lane, Wm1c, s1c, b1c, T2, 128, T1, 128, 64, F1, 32,      64, 2, 2);
  // xyz1: 10(pad32) -> 64 into T1 cols 0..63  => T1 = concat(xyz_enc, feat1)
  mlp_layer(wave, lane, Wx1,  sx1, bx1, Xin, 160, T1, 128, 0,  nullptr, 0,  64, 1, 2);
  // mlp2: 128 -> 128 -> 64 (wq as fp32)
  mlp_layer(wave, lane, Wm2a, s2a, b2a, T1, 128, T2, 128, 0,  nullptr, 0, 128, 4, 2);
  mlp_layer(wave, lane, Wm2b, s2b, b2b, T2, 128, nullptr, 0, 0, WQ, 32,    64, 4, 2);

  // softmax over K=32 per channel + weighted sum of feat1
  if (tid < 64) {
    int co = tid;
    float mx = -3.4e38f;
    for (int k = 0; k < 32; ++k) mx = fmaxf(mx, WQ[co * 32 + k]);
    float ssum = 0.0f, acc = 0.0f;
    for (int k = 0; k < 32; ++k) {
      float e = __expf(WQ[co * 32 + k] - mx);
      ssum += e;
      acc  += e * F1[co * 32 + k];
    }
    pi_feat1[((size_t)b * 64 + co) * S_ + s] = acc / ssum;
  }
}

// ---------------------------------------------------------------------------
// Stage 2: one block per (b,s). Gather K=16 self-neighbors, run
// xyz2 (10->64), mlp3 (192->128->64), softmax over K, weighted sum -> out.
// ---------------------------------------------------------------------------
__global__ void __launch_bounds__(256) stage2_kernel(
    const float* __restrict__ warped_xyz,     // [B][3][S]
    const float* __restrict__ warped_points,  // [B][64][S]
    const float* __restrict__ feat1,          // [B][64][S]
    const int*   __restrict__ idx_s,          // [B][S][16]
    const _Float16* Wx2,  const float* sx2, const float* bx2,
    const _Float16* Wm3a, const float* s3a, const float* b3a,
    const _Float16* Wm3b, const float* s3b, const float* b3b,
    float* __restrict__ out)                  // [B][64][S]
{
  __shared__ _Float16 Xin[16 * 32];    // xyz-feat (10, pad->32, zero-filled)
  __shared__ _Float16 Cbuf[16 * 192];  // concat(xyz_enc, pi_pts, pc_pts_g)
  __shared__ _Float16 T[16 * 128];
  __shared__ float    WP[64 * 16];
  __shared__ float    G[64 * 16];      // gathered feat1 fp32
  __shared__ int      nidx[16];
  __shared__ float    nx[16], ny[16], nz[16];

  const int b = blockIdx.y, s = blockIdx.x;
  const int tid = threadIdx.x, lane = tid & 31, wave = tid >> 5;

  if (tid < 16) {
    int j = idx_s[((size_t)b * S_ + s) * 16 + tid];
    nidx[tid] = j;
    nx[tid] = warped_xyz[(size_t)b * 3 * S_ + j];
    ny[tid] = warped_xyz[(size_t)b * 3 * S_ + S_ + j];
    nz[tid] = warped_xyz[(size_t)b * 3 * S_ + 2 * S_ + j];
  }
  __syncthreads();

  const float qx = warped_xyz[(size_t)b * 3 * S_ + s];
  const float qy = warped_xyz[(size_t)b * 3 * S_ + S_ + s];
  const float qz = warped_xyz[(size_t)b * 3 * S_ + 2 * S_ + s];

  for (int e = tid; e < 16 * 32; e += 256) {
    int k = e >> 5, c = e & 31;
    float v = 0.0f;
    if (c < 3)      v = (c == 0 ? qx : (c == 1 ? qy : qz));
    else if (c < 6) v = (c == 3 ? nx[k] : (c == 4 ? ny[k] : nz[k]));
    else if (c < 9) v = (c == 6 ? nx[k] - qx : (c == 7 ? ny[k] - qy : nz[k] - qz));
    else if (c == 9) {
      float dx = nx[k] - qx, dy = ny[k] - qy, dz = nz[k] - qz;
      v = sqrtf(dx * dx + dy * dy + dz * dz + 1e-20f);
    }
    Xin[e] = (_Float16)v;
  }
  for (int e = tid; e < 16 * 128; e += 256) {
    int k = e >> 7, c = e & 127;
    float v;
    if (c < 64) {
      v = warped_points[((size_t)b * 64 + c) * S_ + s];
    } else {
      v = feat1[((size_t)b * 64 + (c - 64)) * S_ + nidx[k]];
      G[(c - 64) * 16 + k] = v;
    }
    Cbuf[k * 192 + 64 + c] = (_Float16)v;
  }
  __syncthreads();

  // xyz2: 10(pad32) -> 64 into Cbuf cols 0..63
  mlp_layer(wave, lane, Wx2,  sx2, bx2, Xin, 32,  Cbuf, 192, 0, nullptr, 0, 64, 1, 1);
  // mlp3: 192 -> 128 -> 64
  mlp_layer(wave, lane, Wm3a, s3a, b3a, Cbuf, 192, T, 128, 0,  nullptr, 0, 128, 6, 1);
  mlp_layer(wave, lane, Wm3b, s3b, b3b, T, 128, nullptr, 0, 0, WP, 16,      64, 4, 1);

  if (tid < 64) {
    int co = tid;
    float mx = -3.4e38f;
    for (int k = 0; k < 16; ++k) mx = fmaxf(mx, WP[co * 16 + k]);
    float ssum = 0.0f, acc = 0.0f;
    for (int k = 0; k < 16; ++k) {
      float e = __expf(WP[co * 16 + k] - mx);
      ssum += e;
      acc  += e * G[co * 16 + k];
    }
    out[((size_t)b * 64 + co) * S_ + s] = acc / ssum;
  }
}

// ---------------------------------------------------------------------------
// Host driver.  d_in order (sorted-key pytree flatten per layer: W, beta,
// gamma, mean, var):
//  0 warped_xyz  1 warped_points  2 f2_xyz  3 f2_points
//  4..18 mlp1 (3 layers)  19..23 xyz1  24..28 xyz2
//  29..38 mlp2 (2 layers) 39..48 mlp3 (2 layers)
// ---------------------------------------------------------------------------
extern "C" void kernel_launch(void* const* d_in, const int* in_sizes, int n_in,
                              void* d_out, int out_size, void* d_ws, size_t ws_size,
                              hipStream_t stream) {
  (void)in_sizes; (void)n_in; (void)out_size; (void)ws_size;

  const float* warped_xyz    = (const float*)d_in[0];
  const float* warped_points = (const float*)d_in[1];
  const float* f2_xyz        = (const float*)d_in[2];
  const float* f2_points     = (const float*)d_in[3];

  char* ws = (char*)d_ws;
  size_t off = 0;
  auto alloc = [&](size_t bytes) -> char* {
    char* p = ws + off;
    off += (bytes + 255) & ~(size_t)255;
    return p;
  };

  int*   idx_q = (int*)alloc(sizeof(int) * B_ * S_ * KQ);
  int*   idx_s = (int*)alloc(sizeof(int) * B_ * S_ * KS);
  float* feat1 = (float*)alloc(sizeof(float) * B_ * 64 * S_);

  struct Layer { _Float16* W; float* s; float* b; };
  auto packL = [&](int base, int cout, int cin, int cinpad) -> Layer {
    Layer L;
    L.W = (_Float16*)alloc(sizeof(_Float16) * cout * cinpad);
    L.s = (float*)alloc(sizeof(float) * cout);
    L.b = (float*)alloc(sizeof(float) * cout);
    int total = cout * cinpad;
    pack_layer<<<(total + 255) / 256, 256, 0, stream>>>(
        (const float*)d_in[base + 0], (const float*)d_in[base + 1],
        (const float*)d_in[base + 2], (const float*)d_in[base + 3],
        (const float*)d_in[base + 4], cout, cin, cinpad, L.W, L.s, L.b);
    return L;
  };

  Layer m1a = packL(4,  128, 138, 160);
  Layer m1b = packL(9,   64, 128, 128);
  Layer m1c = packL(14,  64,  64,  64);
  Layer x1  = packL(19,  64,  10,  32);
  Layer x2  = packL(24,  64,  10,  32);
  Layer m2a = packL(29, 128, 128, 128);
  Layer m2b = packL(34,  64, 128, 128);
  Layer m3a = packL(39, 128, 192, 192);
  Layer m3b = packL(44,  64, 128, 128);

  dim3 kgrid(S_ / 256, B_);
  knn_kernel<KQ><<<kgrid, 256, 0, stream>>>(warped_xyz, f2_xyz, idx_q, S_, N_);
  knn_kernel<KS><<<kgrid, 256, 0, stream>>>(warped_xyz, warped_xyz, idx_s, S_, S_);

  dim3 g1(S_, B_);
  stage1_kernel<<<g1, 256, 0, stream>>>(
      warped_xyz, warped_points, f2_xyz, f2_points, idx_q,
      m1a.W, m1a.s, m1a.b, m1b.W, m1b.s, m1b.b, m1c.W, m1c.s, m1c.b,
      x1.W,  x1.s,  x1.b,  m2a.W, m2a.s, m2a.b, m2b.W, m2b.s, m2b.b,
      feat1);

  stage2_kernel<<<g1, 256, 0, stream>>>(
      warped_xyz, warped_points, feat1, idx_s,
      x2.W, x2.s, x2.b, m3a.W, m3a.s, m3a.b, m3b.W, m3b.s, m3b.b,
      (float*)d_out);
}
